// OptimizedCrossScaleMambaBlock_71047349010904
// MI455X (gfx1250) — matrix operand order repaired and checked by reading, not compiled, not run on hardware
//
#include <hip/hip_runtime.h>
#include <math.h>

// ---------------- CDNA5 WMMA plumbing (wave32, 16x16x32 f16 -> f32) --------
typedef __attribute__((ext_vector_type(16))) _Float16 v16h;
typedef __attribute__((ext_vector_type(8)))  float    v8f;

__device__ __forceinline__ v8f wmma_f16(v16h a, v16h b, v8f c) {
  // D = A(16x32) * B(32x16) + C ; args: neg_a, A, neg_b, B, c_mod, C, reuse_a, reuse_b
  return __builtin_amdgcn_wmma_f32_16x16x32_f16(false, a, false, b, (short)0, c,
                                                false, false);
}

// A-matrix (16x32) k placement, contiguous order matching B: for K index k2
// (0..31), lane-half = k2 >> 4 (0: lanes 0-15, 1: lanes 16-31), packed
// element = k2 & 15 of the lane's 16-half fragment.
__device__ __forceinline__ int a_half_of_k(int k2) {
  return k2 >> 4;
}
__device__ __forceinline__ int a_elem_of_k(int k2) {
  return k2 & 15;
}
// B-matrix (32x16 KxN): lanes 0-15 hold K=0..15, lanes 16-31 K=16..31,
// element index = k & 15.

__device__ __forceinline__ float silu_f(float x) { return x / (1.f + __expf(-x)); }

#define BN_SCALE 0.99999499f /* rsqrt(1+1e-5) */

// ---------------------------------------------------------------------------
// K1: axial depthwise convs (1x7 -> 7x1 -> 3x3, dilation d) + BN-eval + skip.
// One (b,c) 64x64 plane per block, fully LDS-resident.  Output NHWC.
// ---------------------------------------------------------------------------
__global__ void axial_dw_kernel(const float* __restrict__ x,
                                const float* __restrict__ kh,
                                const float* __restrict__ kw,
                                const float* __restrict__ k3,
                                const float* __restrict__ g,
                                const float* __restrict__ bb,
                                float* __restrict__ y_nhwc, int br, int dil) {
  __shared__ float sin_[4096];
  __shared__ float sA[4096];
  __shared__ float sB[4096];
  int bid = blockIdx.x;
  int b = bid >> 5, c = bid & 31;
  const float* xp = x + ((size_t)(b * 128 + br * 32 + c)) * 4096;
  int t = threadIdx.x;
  for (int i = t; i < 4096; i += 256) sin_[i] = xp[i];
  float wkw[7], wkh[7], wk3[9];
#pragma unroll
  for (int j = 0; j < 7; ++j) { wkw[j] = kw[c * 7 + j]; wkh[j] = kh[c * 7 + j]; }
#pragma unroll
  for (int j = 0; j < 9; ++j) wk3[j] = k3[c * 9 + j];
  __syncthreads();
  for (int i = t; i < 4096; i += 256) {   // 1x7 horizontal
    int h = i >> 6, w = i & 63;
    float acc = 0.f;
#pragma unroll
    for (int j = 0; j < 7; ++j) {
      int ww = w + (j - 3) * dil;
      if (ww >= 0 && ww < 64) acc += sin_[(h << 6) + ww] * wkw[j];
    }
    sA[i] = acc;
  }
  __syncthreads();
  for (int i = t; i < 4096; i += 256) {   // 7x1 vertical
    int h = i >> 6, w = i & 63;
    float acc = 0.f;
#pragma unroll
    for (int j = 0; j < 7; ++j) {
      int hh = h + (j - 3) * dil;
      if (hh >= 0 && hh < 64) acc += sA[(hh << 6) + w] * wkh[j];
    }
    sB[i] = acc;
  }
  __syncthreads();
  for (int i = t; i < 4096; i += 256) {   // 3x3 dilated
    int h = i >> 6, w = i & 63;
    float acc = 0.f;
#pragma unroll
    for (int dy = -1; dy <= 1; ++dy)
#pragma unroll
      for (int dx = -1; dx <= 1; ++dx) {
        int hh = h + dy * dil, ww = w + dx * dil;
        if (hh >= 0 && hh < 64 && ww >= 0 && ww < 64)
          acc += sB[(hh << 6) + ww] * wk3[(dy + 1) * 3 + (dx + 1)];
      }
    sA[i] = acc;
  }
  __syncthreads();
  float gc = g[c] * BN_SCALE, bc = bb[c];
  for (int i = t; i < 4096; i += 256) {
    float yv = sA[i] * gc + bc + sin_[i];
    y_nhwc[((size_t)b * 4096 + i) * 32 + c] = yv;
  }
}

// ---------------------------------------------------------------------------
// K2: LayerNorm(32) + in-proj GEMM [32768,32]x[32,128] via WMMA (K=32 exact).
// Fragments staged in WMMA-native lane layout: operand load = one v16h
// (2x ds_load_b128) per fragment.
// ---------------------------------------------------------------------------
__global__ void inproj_kernel(const float* __restrict__ xin,
                              const float* __restrict__ ln_g,
                              const float* __restrict__ ln_b,
                              const float* __restrict__ in_w,
                              float* __restrict__ xw_nchw,
                              float* __restrict__ zbuf) {
  __shared__ __align__(32) _Float16 AF[8][32][16];  // [wave][lane][elem]
  __shared__ __align__(32) _Float16 BF[8][32][16];  // [ntile][lane][elem]
  int t = threadIdx.x;
  int m0 = blockIdx.x * 128;
  for (int i = t; i < 32 * 128; i += 256) {   // stage W: in_w (128,32)
    int n = i >> 5, k = i & 31;
    BF[n >> 4][(n & 15) + ((k >> 4) << 4)][k & 15] =
        (_Float16)in_w[(size_t)n * 32 + k];
  }
  if (t < 128) {                              // LN + stage A in frag layout
    const float* xr = xin + (size_t)(m0 + t) * 32;
    float s = 0.f, s2 = 0.f;
#pragma unroll
    for (int k = 0; k < 32; ++k) { float v = xr[k]; s += v; s2 += v * v; }
    float mu = s * (1.f / 32.f);
    float var = s2 * (1.f / 32.f) - mu * mu;
    float inv = rsqrtf(var + 1e-5f);
    int wvr = t >> 4, rlo = t & 15;
#pragma unroll
    for (int k = 0; k < 32; ++k) {
      float v = (xr[k] - mu) * inv * ln_g[k] + ln_b[k];
      AF[wvr][rlo + (a_half_of_k(k) << 4)][a_elem_of_k(k)] = (_Float16)v;
    }
  }
  __syncthreads();
  int wv = t >> 5, lane = t & 31, row0 = wv * 16;
  v16h a = *(const v16h*)&AF[wv][lane][0];
  int hi = (lane >> 4) & 1;
#pragma unroll
  for (int nt = 0; nt < 8; ++nt) {
    v16h bf = *(const v16h*)&BF[nt][lane][0];
    v8f c = {};
    c = wmma_f16(a, bf, c);
    int n = nt * 16 + (lane & 15);
#pragma unroll
    for (int v = 0; v < 8; ++v) {
      int grow = m0 + row0 + v + (hi ? 8 : 0);
      int b = grow >> 12, l = grow & 4095;
      if (n < 64)
        xw_nchw[((size_t)b * 64 + n) * 4096 + l] = c[v];
      else
        zbuf[(size_t)grow * 64 + (n - 64)] = c[v];
    }
  }
}

// ---------------------------------------------------------------------------
// K3: depthwise 3x3 (dil 1) + bias + SiLU on xw.  Output NHWC.
// ---------------------------------------------------------------------------
__global__ void dw3_silu_kernel(const float* __restrict__ xw_nchw,
                                const float* __restrict__ cw,
                                const float* __restrict__ cb,
                                float* __restrict__ xc_nhwc) {
  __shared__ float sp[4096];
  int bid = blockIdx.x;
  int b = bid >> 6, c = bid & 63;
  const float* xp = xw_nchw + ((size_t)b * 64 + c) * 4096;
  int t = threadIdx.x;
  for (int i = t; i < 4096; i += 256) sp[i] = xp[i];
  float w9[9];
#pragma unroll
  for (int j = 0; j < 9; ++j) w9[j] = cw[c * 9 + j];
  float bias = cb[c];
  __syncthreads();
  for (int i = t; i < 4096; i += 256) {
    int h = i >> 6, w = i & 63;
    float acc = bias;
#pragma unroll
    for (int dy = -1; dy <= 1; ++dy)
#pragma unroll
      for (int dx = -1; dx <= 1; ++dx) {
        int hh = h + dy, ww = w + dx;
        if (hh >= 0 && hh < 64 && ww >= 0 && ww < 64)
          acc += sp[(hh << 6) + ww] * w9[(dy + 1) * 3 + (dx + 1)];
      }
    xc_nhwc[((size_t)b * 4096 + i) * 64 + c] = silu_f(acc);
  }
}

// ---------------------------------------------------------------------------
// K4: fused 4-direction x-projection GEMM [32768,64]x[64,144] via WMMA.
// P row layout per direction block (36 cols): B[0:16] C[16:32] dt[32:34] pad.
// (keeps scan-side reads 16B-aligned for b128 loads)
// ---------------------------------------------------------------------------
__global__ void xproj_kernel(const float* __restrict__ xc_nhwc,
                             const float* __restrict__ xproj_w,
                             float* __restrict__ Pbuf) {
  __shared__ __align__(32) _Float16 AF[8][2][32][16];  // [wave][kf][lane][elem]
  __shared__ __align__(32) _Float16 BF[9][2][32][16];  // [ntile][kf][lane][elem]
  int t = threadIdx.x;
  int m0 = blockIdx.x * 128;
  for (int i = t; i < 64 * 144; i += 256) {  // stage weights (reordered cols)
    int d = i / 144, col = i % 144;
    int kd = col / 36, j = col % 36;
    // j 0..31 -> x_dbl channel j+2 (B then C); j 32,33 -> channels 0,1 (dt)
    float wv = 0.f;
    if (j < 32)       wv = xproj_w[((size_t)kd * 34 + (j + 2)) * 64 + d];
    else if (j < 34)  wv = xproj_w[((size_t)kd * 34 + (j - 32)) * 64 + d];
    int kf = d >> 5, k2 = d & 31;
    BF[col >> 4][kf][(col & 15) + ((k2 >> 4) << 4)][k2 & 15] = (_Float16)wv;
  }
  for (int i = t; i < 128 * 64; i += 256) {  // stage A
    int r = i >> 6, d = i & 63;
    int kf = d >> 5, k2 = d & 31;
    AF[r >> 4][kf][(r & 15) + (a_half_of_k(k2) << 4)][a_elem_of_k(k2)] =
        (_Float16)xc_nhwc[(size_t)(m0 + r) * 64 + d];
  }
  __syncthreads();
  int wv = t >> 5, lane = t & 31, row0 = wv * 16;
  v16h a0 = *(const v16h*)&AF[wv][0][lane][0];
  v16h a1 = *(const v16h*)&AF[wv][1][lane][0];
  int hi = (lane >> 4) & 1;
#pragma unroll
  for (int nt = 0; nt < 9; ++nt) {
    v16h b0 = *(const v16h*)&BF[nt][0][lane][0];
    v16h b1 = *(const v16h*)&BF[nt][1][lane][0];
    v8f c = {};
    c = wmma_f16(a0, b0, c);
    c = wmma_f16(a1, b1, c);
    int n = nt * 16 + (lane & 15);
#pragma unroll
    for (int v = 0; v < 8; ++v) {
      int grow = m0 + row0 + v + (hi ? 8 : 0);
      Pbuf[(size_t)grow * 144 + n] = c[v];
    }
  }
}

__global__ void zero_kernel(float* __restrict__ p, int n) {
  int i = blockIdx.x * 256 + threadIdx.x;
  if (i < n) p[i] = 0.f;
}

// direction k, step l -> pixel index p
__device__ __forceinline__ int dir_pixel(int k, int l) {
  if (k == 0) return l;
  if (k == 1) return ((l & 63) << 6) | (l >> 6);
  if (k == 2) return 4095 - l;
  int m = 4095 - l;
  return ((m & 63) << 6) | (m >> 6);
}

// ---------------------------------------------------------------------------
// K5: selective scan. Block=(b,k); thread=(d, n-quad): 4 state elems in regs,
// 4096 steps. B/C read as float4 (b128), quad-reduce with wave32 shfl_xor,
// f32 atomics into pixel space, global_prefetch of the next step.
// ---------------------------------------------------------------------------
__global__ void scan_kernel(const float* __restrict__ P,
                            const float* __restrict__ xc,
                            const float* __restrict__ dtw,
                            const float* __restrict__ dtb,
                            const float* __restrict__ A_logs,
                            const float* __restrict__ Ds,
                            float* __restrict__ y_acc) {
  int b = blockIdx.x >> 2, k = blockIdx.x & 3;
  int t = threadIdx.x;
  int d = t >> 2, nq = t & 3, n0 = nq * 4;
  float A0[4], h0[4] = {0.f, 0.f, 0.f, 0.f};
#pragma unroll
  for (int j = 0; j < 4; ++j)
    A0[j] = -__expf(A_logs[((size_t)k * 64 + d) * 16 + n0 + j]);
  float w0 = dtw[((size_t)k * 64 + d) * 2 + 0];
  float w1 = dtw[((size_t)k * 64 + d) * 2 + 1];
  float db = dtb[(size_t)k * 64 + d];
  float Dv = Ds[(size_t)k * 64 + d];
  for (int l = 0; l < 4096; ++l) {
    int p = dir_pixel(k, l);
    if (l + 1 < 4096) {  // global_prefetch_b8 of next step's rows
      int pn = dir_pixel(k, l + 1);
      __builtin_prefetch(P + ((size_t)b * 4096 + pn) * 144 + k * 36, 0, 0);
      __builtin_prefetch(xc + ((size_t)b * 4096 + pn) * 64 + d, 0, 0);
    }
    const float* Pr = P + ((size_t)b * 4096 + p) * 144 + k * 36;
    float4 Bt = *(const float4*)(Pr + n0);        // cols 0..15  : B
    float4 Ct = *(const float4*)(Pr + 16 + n0);   // cols 16..31 : C
    float P0 = Pr[32], P1 = Pr[33];               // delta inputs (broadcast)
    float u = xc[((size_t)b * 4096 + p) * 64 + d];
    float zz = P0 * w0 + P1 * w1 + db;
    float delta = (zz > 20.f) ? zz : log1pf(__expf(zz));  // softplus
    float du = delta * u;
    h0[0] = h0[0] * __expf(delta * A0[0]) + du * Bt.x;
    h0[1] = h0[1] * __expf(delta * A0[1]) + du * Bt.y;
    h0[2] = h0[2] * __expf(delta * A0[2]) + du * Bt.z;
    h0[3] = h0[3] * __expf(delta * A0[3]) + du * Bt.w;
    float y = h0[0] * Ct.x + h0[1] * Ct.y + h0[2] * Ct.z + h0[3] * Ct.w;
    y += __shfl_xor(y, 1, 32);
    y += __shfl_xor(y, 2, 32);
    if (nq == 0)
      atomicAdd(&y_acc[((size_t)b * 4096 + p) * 64 + d], y + Dv * u);
  }
}

// ---------------------------------------------------------------------------
// K6: out-LN + SiLU gate + out-proj GEMM [32768,64]x[64,32] via WMMA, fused
// with VSS residual, per-branch scale, BN-eval + ReLU, stores into d_out.
// ---------------------------------------------------------------------------
__global__ void outproj_kernel(const float* __restrict__ y_acc,
                               const float* __restrict__ zbuf,
                               const float* __restrict__ resid,
                               const float* __restrict__ on_g,
                               const float* __restrict__ on_b,
                               const float* __restrict__ out_w,
                               const float* __restrict__ scale,
                               const float* __restrict__ bn_g,
                               const float* __restrict__ bn_b,
                               float* __restrict__ out, int br) {
  __shared__ __align__(32) _Float16 AF[8][2][32][16];
  __shared__ __align__(32) _Float16 BF[2][2][32][16];
  int t = threadIdx.x, m0 = blockIdx.x * 128;
  for (int i = t; i < 64 * 32; i += 256) {  // stage out_w (32,64)
    int k = i >> 5, n = i & 31;
    int kf = k >> 5, k2 = k & 31;
    BF[n >> 4][kf][(n & 15) + ((k2 >> 4) << 4)][k2 & 15] =
        (_Float16)out_w[(size_t)n * 64 + k];
  }
  if (t < 128) {
    const float* yr = y_acc + (size_t)(m0 + t) * 64;
    const float* zr = zbuf + (size_t)(m0 + t) * 64;
    float s = 0.f, s2 = 0.f;
#pragma unroll
    for (int kk = 0; kk < 64; ++kk) { float v = yr[kk]; s += v; s2 += v * v; }
    float mu = s * (1.f / 64.f);
    float var = s2 * (1.f / 64.f) - mu * mu;
    float inv = rsqrtf(var + 1e-5f);
    int wvr = t >> 4, rlo = t & 15;
#pragma unroll
    for (int kk = 0; kk < 64; ++kk) {
      float yl = (yr[kk] - mu) * inv * on_g[kk] + on_b[kk];
      float v = yl * silu_f(zr[kk]);
      int kf = kk >> 5, k2 = kk & 31;
      AF[wvr][kf][rlo + (a_half_of_k(k2) << 4)][a_elem_of_k(k2)] = (_Float16)v;
    }
  }
  __syncthreads();
  float sc = scale[0];
  int wv = t >> 5, lane = t & 31, row0 = wv * 16;
  v16h a0 = *(const v16h*)&AF[wv][0][lane][0];
  v16h a1 = *(const v16h*)&AF[wv][1][lane][0];
  int hi = (lane >> 4) & 1;
#pragma unroll
  for (int nt = 0; nt < 2; ++nt) {
    v16h b0 = *(const v16h*)&BF[nt][0][lane][0];
    v16h b1 = *(const v16h*)&BF[nt][1][lane][0];
    v8f c = {};
    c = wmma_f16(a0, b0, c);
    c = wmma_f16(a1, b1, c);
    int n = nt * 16 + (lane & 15);
    int cg = br * 32 + n;
    float gbn = bn_g[cg] * BN_SCALE, bbn = bn_b[cg];
#pragma unroll
    for (int v = 0; v < 8; ++v) {
      int grow = m0 + row0 + v + (hi ? 8 : 0);
      int b = grow >> 12, l = grow & 4095;
      float val = (resid[(size_t)grow * 32 + n] + c[v]) * sc;
      val = val * gbn + bbn;
      out[((size_t)b * 128 + cg) * 4096 + l] = fmaxf(val, 0.f);
    }
  }
}

// K7: passthrough branch 4: x4*scale4 -> BN -> ReLU
__global__ void branch4_kernel(const float* __restrict__ x,
                               const float* __restrict__ scale,
                               const float* __restrict__ bn_g,
                               const float* __restrict__ bn_b,
                               float* __restrict__ out) {
  int i = blockIdx.x * 256 + threadIdx.x;  // over 8*32*4096
  int b = i >> 17;
  int c = (i >> 12) & 31;
  int p = i & 4095;
  int cg = 96 + c;
  float v = x[((size_t)b * 128 + cg) * 4096 + p] * scale[0];
  v = v * bn_g[cg] * BN_SCALE + bn_b[cg];
  out[((size_t)b * 128 + cg) * 4096 + p] = fmaxf(v, 0.f);
}

// ---------------------------------------------------------------------------
extern "C" void kernel_launch(void* const* d_in, const int* in_sizes, int n_in,
                              void* d_out, int out_size, void* d_ws,
                              size_t ws_size, hipStream_t stream) {
  (void)in_sizes; (void)n_in; (void)out_size; (void)ws_size;
  const float* x      = (const float*)d_in[0];
  const float* ln_g   = (const float*)d_in[16];
  const float* ln_b   = (const float*)d_in[17];
  const float* in_w   = (const float*)d_in[18];
  const float* conv_w = (const float*)d_in[19];
  const float* conv_b = (const float*)d_in[20];
  const float* xprojw = (const float*)d_in[21];
  const float* dtw    = (const float*)d_in[22];
  const float* dtb    = (const float*)d_in[23];
  const float* A_logs = (const float*)d_in[24];
  const float* Ds     = (const float*)d_in[25];
  const float* on_g   = (const float*)d_in[26];
  const float* on_b   = (const float*)d_in[27];
  const float* out_w  = (const float*)d_in[28];
  const float* bn_g   = (const float*)d_in[33];
  const float* bn_b   = (const float*)d_in[34];
  float* out = (float*)d_out;

  // workspace partition (floats); branches processed sequentially, buffers reused
  float* ws      = (float*)d_ws;
  float* y_nhwc  = ws;                    // 1,048,576  (B*L*32)  VSS input/resid
  float* xw_nchw = y_nhwc + 1048576;      // 2,097,152  (B*64*L)
  float* zbuf    = xw_nchw + 2097152;     // 2,097,152  (B*L*64)
  float* xc_nhwc = zbuf + 2097152;        // 2,097,152  (B*L*64)
  float* Pbuf    = xc_nhwc + 2097152;     // 4,718,592  (B*L*144)
  float* y_acc   = Pbuf + 4718592;        // 2,097,152  (B*L*64)

  for (int br = 0; br < 3; ++br) {
    const float* kh = (const float*)d_in[1 + 5 * br];
    const float* kw = (const float*)d_in[2 + 5 * br];
    const float* k3 = (const float*)d_in[3 + 5 * br];
    const float* dg = (const float*)d_in[4 + 5 * br];
    const float* db = (const float*)d_in[5 + 5 * br];
    const float* sc = (const float*)d_in[29 + br];
    int dil = br + 1;

    axial_dw_kernel<<<256, 256, 0, stream>>>(x, kh, kw, k3, dg, db, y_nhwc, br, dil);
    inproj_kernel<<<256, 256, 0, stream>>>(y_nhwc, ln_g, ln_b, in_w, xw_nchw, zbuf);
    dw3_silu_kernel<<<512, 256, 0, stream>>>(xw_nchw, conv_w, conv_b, xc_nhwc);
    xproj_kernel<<<256, 256, 0, stream>>>(xc_nhwc, xprojw, Pbuf);
    zero_kernel<<<8192, 256, 0, stream>>>(y_acc, 2097152);
    scan_kernel<<<32, 256, 0, stream>>>(Pbuf, xc_nhwc, dtw, dtb, A_logs, Ds, y_acc);
    outproj_kernel<<<256, 256, 0, stream>>>(y_acc, zbuf, y_nhwc, on_g, on_b,
                                            out_w, sc, bn_g, bn_b, out, br);
  }
  branch4_kernel<<<4096, 256, 0, stream>>>(x, (const float*)d_in[32], bn_g, bn_b, out);
}
